// PointNet2_64622077936194
// MI455X (gfx1250) — compile-verified, matrix operand order checked by reading, not run on hardware
//
#include <hip/hip_runtime.h>

// ============================================================================
// PointNet++ (part-seg style) forward for MI455X / gfx1250.
//
// All MLP layers run as WMMA GEMMs (v_wmma_f32_16x16x32_f16).  Activations and
// weights are f16, accumulation and BN statistics f32.  Activation rows are
// stored with stride Kpad = align32(C) and zero-padded tails so the GEMM inner
// loop needs NO guards: each lane's A fragment is two contiguous 16-byte runs
// (global_load_b128 x2) reused across NB output panels; the B panels are
// staged transposed in LDS so each B fragment is two ds_load_b128.
//
// Input order assumption (setup_inputs dict insertion order, params flattened
// module-by-module, each layer as (W, gamma, beta)):
//   d_in[0] = points (16,4096,3) f32
//   d_in[1+3*i] = W_i, d_in[2+3*i] = gamma_i, d_in[3+3*i] = beta_i,
//   i = 0..16 over sa1(3), sa2(3), sa3(3), fp1(2), fp2(2), fp3(3), fc(1).
// ============================================================================

typedef __attribute__((ext_vector_type(16))) _Float16 v16h;
typedef __attribute__((ext_vector_type(8)))  _Float16 v8h;
typedef __attribute__((ext_vector_type(8)))  float    v8f;

#define EPS_BN 1e-5f

union AFrag { v16h v; v8h r[2]; };
union CFrag { v8f  v; float f[8]; };

// ---------------------------------------------------------------------------
// f32 -> f16 copy (weight conversion)
// ---------------------------------------------------------------------------
__global__ void f32_to_f16_kernel(const float* __restrict__ in,
                                  _Float16* __restrict__ out, int n) {
  int i = blockIdx.x * blockDim.x + threadIdx.x;
  if (i < n) out[i] = (_Float16)in[i];
}

// ---------------------------------------------------------------------------
// Farthest point sampling: one block per batch, dist array in LDS.
// Writes gathered centroids (new_xyz) directly.
// ---------------------------------------------------------------------------
__global__ __launch_bounds__(256)
void fps_kernel(const float* __restrict__ xyz, float* __restrict__ new_xyz,
                int B, int N, int S) {
  __shared__ float dist[4096];
  __shared__ float redV[256];
  __shared__ int   redI[256];
  __shared__ float cx, cy, cz;
  const int b = blockIdx.x;
  const int t = threadIdx.x, nt = blockDim.x;
  const float* p = xyz + (size_t)b * N * 3;
  for (int i = t; i < N; i += nt) dist[i] = 1e10f;
  if (t == 0) {
    cx = p[0]; cy = p[1]; cz = p[2];
    new_xyz[(size_t)b * S * 3 + 0] = cx;
    new_xyz[(size_t)b * S * 3 + 1] = cy;
    new_xyz[(size_t)b * S * 3 + 2] = cz;
  }
  __syncthreads();
  for (int it = 1; it < S; ++it) {
    float bestV = -1.f; int bestI = 0;
    for (int i = t; i < N; i += nt) {
      float dx = p[i * 3 + 0] - cx;
      float dy = p[i * 3 + 1] - cy;
      float dz = p[i * 3 + 2] - cz;
      float d = dx * dx + dy * dy + dz * dz;
      float dd = fminf(dist[i], d);
      dist[i] = dd;
      if (dd > bestV) { bestV = dd; bestI = i; }
    }
    redV[t] = bestV; redI[t] = bestI;
    __syncthreads();
    for (int off = nt >> 1; off > 0; off >>= 1) {
      if (t < off && redV[t + off] > redV[t]) {
        redV[t] = redV[t + off]; redI[t] = redI[t + off];
      }
      __syncthreads();
    }
    if (t == 0) {
      int g = redI[0];
      cx = p[g * 3 + 0]; cy = p[g * 3 + 1]; cz = p[g * 3 + 2];
      new_xyz[((size_t)b * S + it) * 3 + 0] = cx;
      new_xyz[((size_t)b * S + it) * 3 + 1] = cy;
      new_xyz[((size_t)b * S + it) * 3 + 2] = cz;
    }
    __syncthreads();
  }
}

// ---------------------------------------------------------------------------
// Ball query: ascending-index scan == "sorted, first nsample" of reference.
// ---------------------------------------------------------------------------
__global__ void ball_query_kernel(const float* __restrict__ xyz,
                                  const float* __restrict__ new_xyz,
                                  int* __restrict__ gidx,
                                  int B, int N, int M, int S, float r2) {
  int gid = blockIdx.x * blockDim.x + threadIdx.x;
  if (gid >= B * M) return;
  int b = gid / M;
  const float* p = xyz + (size_t)b * N * 3;
  float qx = new_xyz[(size_t)gid * 3 + 0];
  float qy = new_xyz[(size_t)gid * 3 + 1];
  float qz = new_xyz[(size_t)gid * 3 + 2];
  int* out = gidx + (size_t)gid * S;
  int cnt = 0, first = -1;
  for (int i = 0; i < N && cnt < S; ++i) {
    float dx = p[i * 3 + 0] - qx;
    float dy = p[i * 3 + 1] - qy;
    float dz = p[i * 3 + 2] - qz;
    float d = dx * dx + dy * dy + dz * dz;
    if (d <= r2) { if (first < 0) first = i; out[cnt++] = i; }
  }
  if (first < 0) first = 0;
  for (; cnt < S; ++cnt) out[cnt] = first;
}

// ---------------------------------------------------------------------------
// Build grouped rows: [xyz - center (3)] ++ [feats (C)] ++ zero pad to Cpad
// ---------------------------------------------------------------------------
__global__ void group_kernel(const float* __restrict__ xyz,
                             const float* __restrict__ new_xyz,
                             const _Float16* __restrict__ feats,
                             const int* __restrict__ gidx,
                             _Float16* __restrict__ out,
                             int B, int N, int M, int S, int C, int Cpad) {
  int row = blockIdx.x * blockDim.x + threadIdx.x;
  if (row >= B * M * S) return;
  int bm = row / S;
  int b = bm / M;
  int g = gidx[row];
  const float* pp = xyz + ((size_t)b * N + g) * 3;
  const float* q  = new_xyz + (size_t)bm * 3;
  _Float16* o = out + (size_t)row * Cpad;
  o[0] = (_Float16)(pp[0] - q[0]);
  o[1] = (_Float16)(pp[1] - q[1]);
  o[2] = (_Float16)(pp[2] - q[2]);
  if (feats) {
    const _Float16* f = feats + ((size_t)b * N + g) * C;
    for (int c = 0; c < C; ++c) o[3 + c] = f[c];
  }
  for (int c = 3 + C; c < Cpad; ++c) o[c] = (_Float16)0.0f;
}

// sa3 rows: [xyz (3, f32)] ++ [feat (C, f16)] ++ pad
__global__ void concat_xyz_feat_kernel(const float* __restrict__ xyz,
                                       const _Float16* __restrict__ f,
                                       _Float16* __restrict__ out,
                                       int rows, int C, int Cpad) {
  int row = blockIdx.x * blockDim.x + threadIdx.x;
  if (row >= rows) return;
  _Float16* o = out + (size_t)row * Cpad;
  o[0] = (_Float16)xyz[(size_t)row * 3 + 0];
  o[1] = (_Float16)xyz[(size_t)row * 3 + 1];
  o[2] = (_Float16)xyz[(size_t)row * 3 + 2];
  const _Float16* fp = f + (size_t)row * C;
  for (int c = 0; c < C; ++c) o[3 + c] = fp[c];
  for (int c = 3 + C; c < Cpad; ++c) o[c] = (_Float16)0.0f;
}

// fp1 rows: [global feat (C1) broadcast over P] ++ [per-point feat (C2)]
__global__ void concat_bcast_kernel(const _Float16* __restrict__ gf,
                                    const _Float16* __restrict__ f,
                                    _Float16* __restrict__ out,
                                    int B, int P, int C1, int C2, int Cpad) {
  int row = blockIdx.x * blockDim.x + threadIdx.x;
  if (row >= B * P) return;
  int b = row / P;
  _Float16* o = out + (size_t)row * Cpad;
  const _Float16* g = gf + (size_t)b * C1;
  for (int c = 0; c < C1; ++c) o[c] = g[c];
  const _Float16* fp = f + (size_t)row * C2;
  for (int c = 0; c < C2; ++c) o[C1 + c] = fp[c];
  for (int c = C1 + C2; c < Cpad; ++c) o[c] = (_Float16)0.0f;
}

// ---------------------------------------------------------------------------
// 3-NN inverse-distance interpolation + concat skip features (+ pad).
// skip16 xor skip32 is non-null.
// ---------------------------------------------------------------------------
__global__ void fp_interp_kernel(const float* __restrict__ unk,
                                 const float* __restrict__ kno,
                                 const _Float16* __restrict__ kf,
                                 const _Float16* __restrict__ skip16,
                                 const float* __restrict__ skip32,
                                 _Float16* __restrict__ out,
                                 int B, int n, int m, int C, int Cs, int Cpad) {
  int gid = blockIdx.x * blockDim.x + threadIdx.x;
  if (gid >= B * n) return;
  int b = gid / n;
  const float* u = unk + (size_t)gid * 3;
  const float* kp = kno + (size_t)b * m * 3;
  float d0 = 1e30f, d1 = 1e30f, d2 = 1e30f;
  int i0 = 0, i1 = 0, i2 = 0;
  for (int j = 0; j < m; ++j) {
    float dx = kp[j * 3 + 0] - u[0];
    float dy = kp[j * 3 + 1] - u[1];
    float dz = kp[j * 3 + 2] - u[2];
    float d = dx * dx + dy * dy + dz * dz;
    if (d < d0)      { d2 = d1; i2 = i1; d1 = d0; i1 = i0; d0 = d; i0 = j; }
    else if (d < d1) { d2 = d1; i2 = i1; d1 = d; i1 = j; }
    else if (d < d2) { d2 = d; i2 = j; }
  }
  float w0 = 1.f / (d0 + 1e-8f);
  float w1 = 1.f / (d1 + 1e-8f);
  float w2 = 1.f / (d2 + 1e-8f);
  float ws = w0 + w1 + w2;
  w0 /= ws; w1 /= ws; w2 /= ws;
  _Float16* o = out + (size_t)gid * Cpad;
  const _Float16* f0 = kf + ((size_t)b * m + i0) * C;
  const _Float16* f1 = kf + ((size_t)b * m + i1) * C;
  const _Float16* f2 = kf + ((size_t)b * m + i2) * C;
  for (int c = 0; c < C; ++c)
    o[c] = (_Float16)(w0 * (float)f0[c] + w1 * (float)f1[c] + w2 * (float)f2[c]);
  if (skip16) {
    const _Float16* s = skip16 + (size_t)gid * Cs;
    for (int c = 0; c < Cs; ++c) o[C + c] = s[c];
  } else {
    const float* s = skip32 + (size_t)gid * Cs;
    for (int c = 0; c < Cs; ++c) o[C + c] = (_Float16)s[c];
  }
  for (int c = C + Cs; c < Cpad; ++c) o[c] = (_Float16)0.0f;
}

// ---------------------------------------------------------------------------
// Max pool over the sample axis: out[bm, c] = max_s act[(bm*S+s), c]
// ---------------------------------------------------------------------------
__global__ void maxpool_kernel(const _Float16* __restrict__ act,
                               _Float16* __restrict__ out,
                               int BM, int S, int C) {
  int gid = blockIdx.x * blockDim.x + threadIdx.x;
  if (gid >= BM * C) return;
  int bm = gid / C, c = gid % C;
  const _Float16* p = act + (size_t)bm * S * C + c;
  float mx = -1e30f;
  for (int s = 0; s < S; ++s) mx = fmaxf(mx, (float)p[(size_t)s * C]);
  out[gid] = (_Float16)mx;
}

// ---------------------------------------------------------------------------
// WMMA GEMM + BN-statistics:  Y[rows,N] = A[rows,Kpad] @ W[K,N]
//   Block tile: 64 rows x (NB*16) cols; 4 waves along M; each wave reuses its
//   A fragment (2 x global_load_b128) across NB WMMAs per 32-wide k-chunk.
//   B panels staged TRANSPOSED in LDS: ldsB[col*Kpad + k] -> ds_load_b128 x2.
// A-fragment layout per CDNA5 ISA (16-bit A 16x32):
//   lanes 0-15 : row M=lane,    K = {0..7, 16..23}
//   lanes 16-31: row M=lane-16, K = {8..15, 24..31}
// Per-channel sum / sum-of-squares reduced in LDS, one global atomic each.
// ---------------------------------------------------------------------------
template <int NB>
__global__ __launch_bounds__(128)
void gemm_bn_stats_kernel(const _Float16* __restrict__ A,
                          const _Float16* __restrict__ W,
                          float* __restrict__ Y,
                          float* __restrict__ stats,   // [2*N]: sums, sumsq
                          int rows, int K, int N, int Kpad) {
  extern __shared__ _Float16 ldsB[];          // NCOLS * Kpad, layout [col][k]
  constexpr int NCOLS = NB * 16;
  constexpr int CSH   = (NB == 4) ? 6 : ((NB == 2) ? 5 : 4);
  const int nTile = blockIdx.y * NCOLS;

  // cooperative, coalesced load of the weight panels (zero-padded K),
  // transposed into [col][k] layout
  for (int i = threadIdx.x; i < Kpad * NCOLS; i += blockDim.x) {
    int k = i >> CSH, c = i & (NCOLS - 1);
    ldsB[c * Kpad + k] = (k < K) ? W[(size_t)k * N + nTile + c] : (_Float16)0.0f;
  }
  __shared__ float sSum[NCOLS], sSqs[NCOLS];
  for (int i = threadIdx.x; i < NCOLS; i += blockDim.x) {
    sSum[i] = 0.f; sSqs[i] = 0.f;
  }
  __syncthreads();

  const int lane  = threadIdx.x & 31;
  const int wave  = threadIdx.x >> 5;
  const int mTile = (blockIdx.x * 4 + wave) * 16;
  const int row   = mTile + (lane & 15);
  const int kHalf = (lane >> 4) * 8;          // 0 or 8
  const int nCol  = lane & 15;

  CFrag acc[NB];
  #pragma unroll
  for (int nt = 0; nt < NB; ++nt)
    #pragma unroll
    for (int r = 0; r < 8; ++r) acc[nt].f[r] = 0.f;

  // both pointers are 16-byte aligned: base 256B-aligned, row stride Kpad*2
  // (mult of 64B), kHalf*2 in {0,16}
  const v8h* aPtr = (const v8h*)(A + (size_t)row * Kpad + kHalf);
  const v8h* bPtr[NB];
  #pragma unroll
  for (int nt = 0; nt < NB; ++nt)
    bPtr[nt] = (const v8h*)(ldsB + (size_t)(nt * 16 + nCol) * Kpad + kHalf);

  const int nIter = Kpad >> 5;
  for (int i = 0; i < nIter; ++i) {
    __builtin_prefetch((const _Float16*)(aPtr + i * 4) + 64, 0, 3);
    AFrag a;
    a.r[0] = aPtr[i * 4];           // K = k0 + kHalf + 0..7
    a.r[1] = aPtr[i * 4 + 2];       // K = k0 + kHalf + 16..23
    #pragma unroll
    for (int nt = 0; nt < NB; ++nt) {
      AFrag b;
      b.r[0] = bPtr[nt][i * 4];
      b.r[1] = bPtr[nt][i * 4 + 2];
      acc[nt].v = __builtin_amdgcn_wmma_f32_16x16x32_f16(
          false, a.v, false, b.v, (short)0, acc[nt].v, false, false);
    }
  }

  // store D tiles + per-channel partial sums
  const int mOff = (lane >> 4) * 8;
  #pragma unroll
  for (int nt = 0; nt < NB; ++nt) {
    float s = 0.f, s2 = 0.f;
    #pragma unroll
    for (int r = 0; r < 8; ++r) {
      float v = acc[nt].f[r];
      Y[(size_t)(mTile + mOff + r) * N + nTile + nt * 16 + nCol] = v;
      s += v; s2 += v * v;
    }
    atomicAdd(&sSum[nt * 16 + nCol], s);
    atomicAdd(&sSqs[nt * 16 + nCol], s2);
  }
  __syncthreads();
  for (int i = threadIdx.x; i < NCOLS; i += blockDim.x) {
    atomicAdd(&stats[nTile + i],     sSum[i]);
    atomicAdd(&stats[N + nTile + i], sSqs[i]);
  }
}

// ---------------------------------------------------------------------------
// Apply BN + ReLU, write f16 activations
// ---------------------------------------------------------------------------
__global__ void bn_relu_kernel(const float* __restrict__ Y,
                               const float* __restrict__ stats,
                               const float* __restrict__ gamma,
                               const float* __restrict__ beta,
                               _Float16* __restrict__ out,
                               long long total, int rows, int N) {
  long long gid = (long long)blockIdx.x * blockDim.x + threadIdx.x;
  if (gid >= total) return;
  int c = (int)(gid % N);
  float inv  = 1.f / (float)rows;
  float mean = stats[c] * inv;
  float var  = stats[N + c] * inv - mean * mean;
  float v = gamma[c] * (Y[gid] - mean) * rsqrtf(var + EPS_BN) + beta[c];
  out[gid] = (_Float16)fmaxf(v, 0.f);
}

// Final layer: BN + ReLU + transpose (B, Npts, C) -> (B, C, Npts), f32 out
__global__ void bn_relu_transpose_kernel(const float* __restrict__ Y,
                                         const float* __restrict__ stats,
                                         const float* __restrict__ gamma,
                                         const float* __restrict__ beta,
                                         float* __restrict__ out,
                                         int B, int Npts, int C) {
  long long total = (long long)B * Npts * C;
  long long gid = (long long)blockIdx.x * blockDim.x + threadIdx.x;
  if (gid >= total) return;
  int c = (int)(gid % C);
  long long row = gid / C;
  int b  = (int)(row / Npts);
  int nn = (int)(row % Npts);
  float inv  = 1.f / (float)(B * Npts);
  float mean = stats[c] * inv;
  float var  = stats[C + c] * inv - mean * mean;
  float v = gamma[c] * (Y[gid] - mean) * rsqrtf(var + EPS_BN) + beta[c];
  out[((long long)b * C + c) * Npts + nn] = fmaxf(v, 0.f);
}

// ============================================================================
// Host driver
// ============================================================================
static const int WIN[17]  = {3, 64, 64,   131, 128, 128,  259, 256, 512,
                             1280, 256,   384, 256,       131, 128, 128, 128};
static const int WOUT[17] = {64, 64, 128, 128, 128, 256,  256, 512, 1024,
                             256, 256,    256, 128,       128, 128, 128, 128};

static inline int align32(int x) { return (x + 31) & ~31; }

extern "C" void kernel_launch(void* const* d_in, const int* in_sizes, int n_in,
                              void* d_out, int out_size, void* d_ws, size_t ws_size,
                              hipStream_t stream) {
  (void)in_sizes; (void)n_in; (void)out_size; (void)ws_size;

  const int B = 16, N0 = 4096, S1 = 512, NS1 = 64, S2 = 128, NS2 = 64;
  const float R1sq = 0.2f * 0.2f, R2sq = 0.4f * 0.4f;
  const int rows1 = B * S1 * NS1;   // 524288
  const int rows2 = B * S2 * NS2;   // 131072
  const int rows3 = B * S2;         // 2048
  const int rowsF2 = B * S1;        // 8192
  const int rowsF3 = B * N0;        // 65536

  const float* points = (const float*)d_in[0];

  // ---- workspace carve-up (assumed large enough; ~545 MB total) ----
  char* wsb = (char*)d_ws;
  size_t off = 0;
  auto carve = [&](size_t bytes) -> void* {
    off = (off + 255) & ~(size_t)255;
    void* p = wsb + off;
    off += bytes;
    return p;
  };

  _Float16* w16[17];
  {
    size_t tot = 0;
    for (int i = 0; i < 17; ++i) tot += (size_t)WIN[i] * WOUT[i];
    _Float16* base = (_Float16*)carve(tot * sizeof(_Float16));
    size_t o = 0;
    for (int i = 0; i < 17; ++i) { w16[i] = base + o; o += (size_t)WIN[i] * WOUT[i]; }
  }
  float* stats = (float*)carve(2 * 1024 * sizeof(float));
  float* nx1   = (float*)carve((size_t)B * S1 * 3 * sizeof(float));
  float* nx2   = (float*)carve((size_t)B * S2 * 3 * sizeof(float));
  int*   g1    = (int*)carve((size_t)rows1 * sizeof(int));
  int*   g2    = (int*)carve((size_t)rows2 * sizeof(int));
  _Float16* l1f  = (_Float16*)carve((size_t)B * S1 * 128 * sizeof(_Float16));
  _Float16* l2f  = (_Float16*)carve((size_t)B * S2 * 256 * sizeof(_Float16));
  _Float16* l3f  = (_Float16*)carve((size_t)B * 1024 * sizeof(_Float16));
  _Float16* l2fp = (_Float16*)carve((size_t)B * S2 * 256 * sizeof(_Float16));
  _Float16* l1fp = (_Float16*)carve((size_t)B * S1 * 128 * sizeof(_Float16));
  _Float16* bA   = (_Float16*)carve((size_t)rows1 * 128 * sizeof(_Float16));
  _Float16* bB   = (_Float16*)carve((size_t)rows1 * 128 * sizeof(_Float16));
  float*    Ybuf = (float*)carve((size_t)rows1 * 128 * sizeof(float));

  // ---- convert all weights to f16 ----
  for (int i = 0; i < 17; ++i) {
    int n = WIN[i] * WOUT[i];
    f32_to_f16_kernel<<<(n + 255) / 256, 256, 0, stream>>>(
        (const float*)d_in[1 + 3 * i], w16[i], n);
  }

  // ---- GEMM + BN + ReLU layer helper ----
  // NB (output panels per block): dynamic LDS = NB*16*Kpad*2: <= 32KB for Kpad <= 512, 40KB at Kpad 1280.
  auto gemm_bn = [&](const _Float16* A, int rows, int li, _Float16* out16) {
    int K = WIN[li], N = WOUT[li];
    int Kpad = align32(K);
    hipMemsetAsync(stats, 0, 2 * N * sizeof(float), stream);
    if (Kpad <= 256) {
      dim3 grid(rows / 64, N / 64);
      size_t lds = (size_t)Kpad * 64 * sizeof(_Float16);
      gemm_bn_stats_kernel<4><<<grid, 128, lds, stream>>>(A, w16[li], Ybuf,
                                                          stats, rows, K, N, Kpad);
    } else if (Kpad <= 512) {
      dim3 grid(rows / 64, N / 32);
      size_t lds = (size_t)Kpad * 32 * sizeof(_Float16);
      gemm_bn_stats_kernel<2><<<grid, 128, lds, stream>>>(A, w16[li], Ybuf,
                                                          stats, rows, K, N, Kpad);
    } else {
      dim3 grid(rows / 64, N / 16);
      size_t lds = (size_t)Kpad * 16 * sizeof(_Float16);
      gemm_bn_stats_kernel<1><<<grid, 128, lds, stream>>>(A, w16[li], Ybuf,
                                                          stats, rows, K, N, Kpad);
    }
    long long total = (long long)rows * N;
    bn_relu_kernel<<<(int)((total + 255) / 256), 256, 0, stream>>>(
        Ybuf, stats, (const float*)d_in[2 + 3 * li],
        (const float*)d_in[3 + 3 * li], out16, total, rows, N);
  };

  // ================= SA1 =================
  fps_kernel<<<B, 256, 0, stream>>>(points, nx1, B, N0, S1);
  ball_query_kernel<<<(B * S1 + 255) / 256, 256, 0, stream>>>(
      points, nx1, g1, B, N0, S1, NS1, R1sq);
  group_kernel<<<(rows1 + 255) / 256, 256, 0, stream>>>(
      points, nx1, (const _Float16*)nullptr, g1, bA,
      B, N0, S1, NS1, 0, align32(3));
  gemm_bn(bA, rows1, 0, bB);
  gemm_bn(bB, rows1, 1, bA);
  gemm_bn(bA, rows1, 2, bB);
  maxpool_kernel<<<(B * S1 * 128 + 255) / 256, 256, 0, stream>>>(
      bB, l1f, B * S1, NS1, 128);

  // ================= SA2 =================
  fps_kernel<<<B, 256, 0, stream>>>(nx1, nx2, B, S1, S2);
  ball_query_kernel<<<(B * S2 + 255) / 256, 256, 0, stream>>>(
      nx1, nx2, g2, B, S1, S2, NS2, R2sq);
  group_kernel<<<(rows2 + 255) / 256, 256, 0, stream>>>(
      nx1, nx2, l1f, g2, bA, B, S1, S2, NS2, 128, align32(131));
  gemm_bn(bA, rows2, 3, bB);
  gemm_bn(bB, rows2, 4, bA);
  gemm_bn(bA, rows2, 5, bB);
  maxpool_kernel<<<(B * S2 * 256 + 255) / 256, 256, 0, stream>>>(
      bB, l2f, B * S2, NS2, 256);

  // ================= SA3 (global) =================
  concat_xyz_feat_kernel<<<(rows3 + 255) / 256, 256, 0, stream>>>(
      nx2, l2f, bA, rows3, 256, align32(259));
  gemm_bn(bA, rows3, 6, bB);
  gemm_bn(bB, rows3, 7, bA);
  gemm_bn(bA, rows3, 8, bB);
  maxpool_kernel<<<(B * 1024 + 255) / 256, 256, 0, stream>>>(
      bB, l3f, B, S2, 1024);

  // ================= FP1 (broadcast interp) =================
  concat_bcast_kernel<<<(rows3 + 255) / 256, 256, 0, stream>>>(
      l3f, l2f, bA, B, S2, 1024, 256, align32(1280));
  gemm_bn(bA, rows3, 9, bB);
  gemm_bn(bB, rows3, 10, l2fp);

  // ================= FP2 (3-NN interp 512 <- 128) =================
  fp_interp_kernel<<<(B * S1 + 255) / 256, 256, 0, stream>>>(
      nx1, nx2, l2fp, l1f, (const float*)nullptr, bA,
      B, S1, S2, 256, 128, align32(384));
  gemm_bn(bA, rowsF2, 11, bB);
  gemm_bn(bB, rowsF2, 12, l1fp);

  // ================= FP3 (3-NN interp 4096 <- 512) =================
  fp_interp_kernel<<<(B * N0 + 255) / 256, 256, 0, stream>>>(
      points, nx1, l1fp, (const _Float16*)nullptr, points, bA,
      B, N0, S1, 128, 3, align32(131));
  gemm_bn(bA, rowsF3, 13, bB);
  gemm_bn(bB, rowsF3, 14, bA);
  gemm_bn(bA, rowsF3, 15, bB);

  // ================= FC + BN + ReLU + transpose =================
  {
    int li = 16, K = 128, N = 128, Kpad = align32(K);
    hipMemsetAsync(stats, 0, 2 * N * sizeof(float), stream);
    dim3 grid(rowsF3 / 64, N / 64);
    size_t lds = (size_t)Kpad * 64 * sizeof(_Float16);
    gemm_bn_stats_kernel<4><<<grid, 128, lds, stream>>>(bB, w16[li], Ybuf,
                                                        stats, rowsF3, K, N, Kpad);
    long long total = (long long)rowsF3 * N;
    bn_relu_transpose_kernel<<<(int)((total + 255) / 256), 256, 0, stream>>>(
        Ybuf, stats, (const float*)d_in[2 + 3 * li],
        (const float*)d_in[3 + 3 * li], (float*)d_out, B, N0, N);
  }
}
